// SqueezeBertModule_46746424049861
// MI455X (gfx1250) — matrix operand order repaired and checked by reading, not compiled, not run on hardware
//
#include <hip/hip_runtime.h>

// ---------------------------------------------------------------------------
// CDNA5 (gfx1250) SqueezeBERT layer: bf16 WMMA GEMMs + flash attention.
// wave32; V_WMMA_F32_16X16X32_BF16 everywhere for matrix math.
// ---------------------------------------------------------------------------

typedef __attribute__((ext_vector_type(16))) __bf16 v16bf;
typedef __attribute__((ext_vector_type(8)))  float  v8f;

#define BB 16
#define CC 768
#define SS 1024
#define HH 12
#define DD 64
#define II 3072

__device__ __forceinline__ unsigned int pack2_bf16(float a, float b) {
    unsigned int ua = __float_as_uint(a);
    unsigned int ub = __float_as_uint(b);
    ua = (ua + 0x7FFFu + ((ua >> 16) & 1u)) >> 16;   // RNE
    ub = (ub + 0x7FFFu + ((ub >> 16) & 1u)) >> 16;
    return ua | (ub << 16);
}

__device__ __forceinline__ unsigned short bf16_bits(float a) {
    unsigned int ua = __float_as_uint(a);
    return (unsigned short)((ua + 0x7FFFu + ((ua >> 16) & 1u)) >> 16);
}

union FragU { uint4 u[2]; v16bf v; };

__device__ __forceinline__ v8f v8f_zero() {
    v8f z;
#pragma unroll
    for (int i = 0; i < 8; ++i) z[i] = 0.f;
    return z;
}

// A fragment (16x32 bf16): LDS layout [m][k], krow = bf16 elems per row.
// lane: m = lane%16, kbase = (lane/16)*8; VGPR0-3 <- K kbase..kbase+7,
// VGPR4-7 <- K 16+kbase..+7 (pairs packed lo/hi).
__device__ __forceinline__ v16bf load_frag_a(const unsigned int* sm, int mbase,
                                             int krow, int koff) {
    const int lane = threadIdx.x & 31;
    const int kb = (lane >> 4) << 3;
    const int m  = mbase + (lane & 15);
    const uint4* p = (const uint4*)sm;
    const int e0 = m * krow + koff + kb;          // multiple of 8 -> 16B aligned
    FragU f;
    f.u[0] = p[e0 >> 3];
    f.u[1] = p[(e0 + 16) >> 3];
    return f.v;
}

// B fragment (32x16 bf16): LDS layout [n][k].
// lane: n = lane%16, kbase = (lane/16)*8; VGPR0-3 <- K kbase..kbase+7,
// VGPR4-7 <- K 16+kbase..+7 (pairs packed lo/hi).
__device__ __forceinline__ v16bf load_frag_b(const unsigned int* sm, int nbase,
                                             int krow, int koff) {
    const int lane = threadIdx.x & 31;
    const int kb = (lane >> 4) << 3;
    const int n  = nbase + (lane & 15);
    const uint4* p = (const uint4*)sm;
    const int e0 = n * krow + koff + kb;
    FragU f;
    f.u[0] = p[e0 >> 3];
    f.u[1] = p[(e0 + 16) >> 3];
    return f.v;
}

#define WMMA_BF16(A, Bf, Cacc) \
    __builtin_amdgcn_wmma_f32_16x16x32_bf16(false, (A), false, (Bf), (short)0, (Cacc), false, false)

// ---------------------------------------------------------------------------
// Generic NCW 1x1-conv GEMM: out[z,m,n] = sum_k W[m,k] * X[z,k,n] + bias[m]
// (+ optional residual, optional exact GELU). Macro-tile 128Mx128N, K chunk 32.
// 8 waves: wave = (wm<2bits? ) -> wm = wave>>1 owns 32 M rows, wn = wave&1 owns 64 N cols.
// ---------------------------------------------------------------------------
template <bool EP_RES, bool EP_GELU>
__global__ __launch_bounds__(256) void wmma_gemm_nchw(
    const float* __restrict__ X, const float* __restrict__ W,
    const float* __restrict__ bias, const float* __restrict__ res,
    float* __restrict__ out, int M, int N, int K)
{
    __shared__ unsigned int sA[128 * 32 / 2];   // [m][k] bf16  (8 KB)
    __shared__ unsigned int sB[128 * 32 / 2];   // [n][k] bf16  (8 KB)

    const int t = threadIdx.x;
    const int lane = t & 31, wave = t >> 5;
    const int n0 = blockIdx.x * 128, m0 = blockIdx.y * 128;
    const size_t zb = blockIdx.z;
    const float* Xb = X + zb * (size_t)K * N;

    v8f acc[2][4];
#pragma unroll
    for (int i = 0; i < 2; ++i)
#pragma unroll
        for (int j = 0; j < 4; ++j) acc[i][j] = v8f_zero();

    const int wm = wave >> 1, wn = wave & 1;

    for (int kc = 0; kc < K; kc += 32) {
        // ---- stage A (weights, row-major [m][k], straight copy) ----
        {
            const int m = t >> 1, ks = (t & 1) << 4;
            const float4* g = (const float4*)(W + (size_t)(m0 + m) * K + kc + ks);
            float4 f0 = g[0], f1 = g[1], f2 = g[2], f3 = g[3];
            uint4 u0 = make_uint4(pack2_bf16(f0.x, f0.y), pack2_bf16(f0.z, f0.w),
                                  pack2_bf16(f1.x, f1.y), pack2_bf16(f1.z, f1.w));
            uint4 u1 = make_uint4(pack2_bf16(f2.x, f2.y), pack2_bf16(f2.z, f2.w),
                                  pack2_bf16(f3.x, f3.y), pack2_bf16(f3.z, f3.w));
            uint4* s4 = (uint4*)sA;
            s4[m * 4 + (ks >> 3)]     = u0;
            s4[m * 4 + (ks >> 3) + 1] = u1;
            if (kc + 32 < K)
                __builtin_prefetch(W + (size_t)(m0 + m) * K + kc + 32 + ks, 0, 3);
        }
        // ---- stage B (activations, transpose [k][n] -> [n][k]) ----
        {
            const int k2 = (t & 15) << 1, ns = (t >> 4) << 3;
            const float4* g0 = (const float4*)(Xb + (size_t)(kc + k2) * N + n0 + ns);
            const float4* g1 = (const float4*)(Xb + (size_t)(kc + k2 + 1) * N + n0 + ns);
            float4 a0 = g0[0], a1 = g0[1], b0 = g1[0], b1 = g1[1];
            float r0[8] = {a0.x, a0.y, a0.z, a0.w, a1.x, a1.y, a1.z, a1.w};
            float r1[8] = {b0.x, b0.y, b0.z, b0.w, b1.x, b1.y, b1.z, b1.w};
#pragma unroll
            for (int j = 0; j < 8; ++j)
                sB[(ns + j) * 16 + (k2 >> 1)] = pack2_bf16(r0[j], r1[j]);
            if (kc + 32 < K)
                __builtin_prefetch(Xb + (size_t)(kc + 32 + k2) * N + n0 + ns, 0, 3);
        }
        __syncthreads();

        v16bf a0 = load_frag_a(sA, wm * 32,      32, 0);
        v16bf a1 = load_frag_a(sA, wm * 32 + 16, 32, 0);
#pragma unroll
        for (int j = 0; j < 4; ++j) {
            v16bf bf = load_frag_b(sB, wn * 64 + j * 16, 32, 0);
            acc[0][j] = WMMA_BF16(a0, bf, acc[0][j]);
            acc[1][j] = WMMA_BF16(a1, bf, acc[1][j]);
        }
        __syncthreads();
    }

    // ---- epilogue: bias (+res) (+gelu), C/D layout: vgpr r, lane l ->
    // (m = r + 8*(l/16), n = l%16) ----
    const int nl = lane & 15, mh = (lane >> 4) << 3;
#pragma unroll
    for (int i = 0; i < 2; ++i)
#pragma unroll
        for (int j = 0; j < 4; ++j) {
            const int mb = m0 + wm * 32 + i * 16 + mh;
            const int nb = n0 + wn * 64 + j * 16 + nl;
#pragma unroll
            for (int r = 0; r < 8; ++r) {
                const int m = mb + r;
                float v = acc[i][j][r] + bias[m];
                const size_t gi = zb * (size_t)M * N + (size_t)m * N + nb;
                if (EP_RES)  v += res[gi];
                if (EP_GELU) v = 0.5f * v * (1.0f + erff(v * 0.70710678118f));
                out[gi] = v;
            }
        }
}

// ---------------------------------------------------------------------------
// Flash attention. grid = (S/64, H, B), block = 128 (4 waves).
// Q/K/V are conv outputs in NCW [B][C][S]; per (b,h) a [D=64][S] slice.
// Each wave owns 16 q-rows; online softmax over 64-wide k blocks.
// ---------------------------------------------------------------------------
__global__ __launch_bounds__(128) void attn_flash(
    const float* __restrict__ qg, const float* __restrict__ kg,
    const float* __restrict__ vg, const float* __restrict__ mask,
    float* __restrict__ ctx)
{
    __shared__ unsigned int sQ[64 * 64 / 2];   // [q][d] bf16 (8 KB)
    __shared__ unsigned int sK[64 * 64 / 2];   // [k][d] bf16 (8 KB)
    __shared__ unsigned int sV[64 * 64 / 2];   // [d][k] bf16 (8 KB)
    __shared__ unsigned int sP[4 * 512];       // per-wave [16 q][64 k] bf16 (8 KB)

    const int t = threadIdx.x;
    const int lane = t & 31, wave = t >> 5;
    const int q0 = blockIdx.x * 64;
    const int h  = blockIdx.y;
    const size_t b = blockIdx.z;

    const float* Q = qg + (b * CC + (size_t)h * DD) * SS;
    const float* Kp = kg + (b * CC + (size_t)h * DD) * SS;
    const float* Vp = vg + (b * CC + (size_t)h * DD) * SS;

    // ---- stage Q: transpose [d][q] -> [q][d] bf16; pair d's to pack dwords ----
    {
        const int d2 = (t & 31) << 1, qs = (t >> 5) << 4;
        const float4* g0 = (const float4*)(Q + (size_t)d2 * SS + q0 + qs);
        const float4* g1 = (const float4*)(Q + (size_t)(d2 + 1) * SS + q0 + qs);
#pragma unroll
        for (int j = 0; j < 4; ++j) {
            float4 x0 = g0[j], x1 = g1[j];
            const int qq = qs + j * 4;
            sQ[(qq + 0) * 32 + (d2 >> 1)] = pack2_bf16(x0.x, x1.x);
            sQ[(qq + 1) * 32 + (d2 >> 1)] = pack2_bf16(x0.y, x1.y);
            sQ[(qq + 2) * 32 + (d2 >> 1)] = pack2_bf16(x0.z, x1.z);
            sQ[(qq + 3) * 32 + (d2 >> 1)] = pack2_bf16(x0.w, x1.w);
        }
    }

    v8f cacc[4];
#pragma unroll
    for (int j = 0; j < 4; ++j) cacc[j] = v8f_zero();
    float mrow[8], lrow[8];
#pragma unroll
    for (int r = 0; r < 8; ++r) { mrow[r] = -1e30f; lrow[r] = 0.f; }

    const int nl = lane & 15, mh = (lane >> 4) << 3;

    for (int kblk = 0; kblk < SS; kblk += 64) {
        __syncthreads();
        // ---- stage K: transpose [d][k] -> [k][d] ----
        {
            const int d2 = (t & 31) << 1, ks = (t >> 5) << 4;
            const float4* g0 = (const float4*)(Kp + (size_t)d2 * SS + kblk + ks);
            const float4* g1 = (const float4*)(Kp + (size_t)(d2 + 1) * SS + kblk + ks);
#pragma unroll
            for (int j = 0; j < 4; ++j) {
                float4 x0 = g0[j], x1 = g1[j];
                const int kk = ks + j * 4;
                sK[(kk + 0) * 32 + (d2 >> 1)] = pack2_bf16(x0.x, x1.x);
                sK[(kk + 1) * 32 + (d2 >> 1)] = pack2_bf16(x0.y, x1.y);
                sK[(kk + 2) * 32 + (d2 >> 1)] = pack2_bf16(x0.z, x1.z);
                sK[(kk + 3) * 32 + (d2 >> 1)] = pack2_bf16(x0.w, x1.w);
            }
        }
        // ---- stage V: straight copy [d][k] ----
        {
            const int d = t >> 1, ks = (t & 1) << 5;
            const float4* g = (const float4*)(Vp + (size_t)d * SS + kblk + ks);
            uint4* s4 = (uint4*)sV;
#pragma unroll
            for (int j = 0; j < 2; ++j) {
                float4 x0 = g[4 * j + 0], x1 = g[4 * j + 1];
                float4 x2 = g[4 * j + 2], x3 = g[4 * j + 3];
                uint4 u0 = make_uint4(pack2_bf16(x0.x, x0.y), pack2_bf16(x0.z, x0.w),
                                      pack2_bf16(x1.x, x1.y), pack2_bf16(x1.z, x1.w));
                uint4 u1 = make_uint4(pack2_bf16(x2.x, x2.y), pack2_bf16(x2.z, x2.w),
                                      pack2_bf16(x3.x, x3.y), pack2_bf16(x3.z, x3.w));
                s4[d * 8 + (ks >> 3) + 2 * j]     = u0;
                s4[d * 8 + (ks >> 3) + 2 * j + 1] = u1;
            }
        }
        __syncthreads();

        // ---- S = Q * K^T  (16q x 64k per wave) ----
        v8f sc[4];
#pragma unroll
        for (int j = 0; j < 4; ++j) sc[j] = v8f_zero();
#pragma unroll
        for (int ch = 0; ch < 64; ch += 32) {
            v16bf a = load_frag_a(sQ, wave * 16, 64, ch);
#pragma unroll
            for (int j = 0; j < 4; ++j) {
                v16bf bf = load_frag_b(sK, j * 16, 64, ch);
                sc[j] = WMMA_BF16(a, bf, sc[j]);
            }
        }
        // ---- scale + mask ----
#pragma unroll
        for (int j = 0; j < 4; ++j) {
            const float mj = mask[b * SS + kblk + j * 16 + nl];
#pragma unroll
            for (int r = 0; r < 8; ++r) sc[j][r] = sc[j][r] * 0.125f + mj;
        }
        // ---- online softmax: per-row (r) stats, 16-lane butterfly reduce ----
#pragma unroll
        for (int r = 0; r < 8; ++r) {
            float v = fmaxf(fmaxf(sc[0][r], sc[1][r]), fmaxf(sc[2][r], sc[3][r]));
#pragma unroll
            for (int off = 8; off >= 1; off >>= 1) v = fmaxf(v, __shfl_xor(v, off, 16));
            const float mn = fmaxf(mrow[r], v);
            const float alpha = __expf(mrow[r] - mn);
            mrow[r] = mn;
            lrow[r] *= alpha;
#pragma unroll
            for (int jd = 0; jd < 4; ++jd) cacc[jd][r] *= alpha;
            float ps = 0.f;
#pragma unroll
            for (int j = 0; j < 4; ++j) {
                const float e = __expf(sc[j][r] - mn);
                sc[j][r] = e;
                ps += e;
            }
#pragma unroll
            for (int off = 8; off >= 1; off >>= 1) ps += __shfl_xor(ps, off, 16);
            lrow[r] += ps;
        }
        // ---- P (C/D layout) -> per-wave LDS in A-fragment layout [q][k] ----
        {
            unsigned short* ph = (unsigned short*)sP + wave * 1024;
#pragma unroll
            for (int j = 0; j < 4; ++j)
#pragma unroll
                for (int r = 0; r < 8; ++r)
                    ph[(r + mh) * 64 + j * 16 + nl] = bf16_bits(sc[j][r]);
        }
        asm volatile("s_wait_dscnt 0" ::: "memory");   // wave-local LDS RAW
        // ---- ctx += P * V ----
#pragma unroll
        for (int ch = 0; ch < 64; ch += 32) {
            v16bf a = load_frag_a(sP + wave * 512, 0, 64, ch);
#pragma unroll
            for (int jd = 0; jd < 4; ++jd) {
                v16bf bf = load_frag_b(sV, jd * 16, 64, ch);
                cacc[jd] = WMMA_BF16(a, bf, cacc[jd]);
            }
        }
    }

    // ---- write ctx back in NCW: c = h*64 + d (n), s = q (m, contiguous) ----
#pragma unroll
    for (int jd = 0; jd < 4; ++jd) {
        float o[8];
#pragma unroll
        for (int r = 0; r < 8; ++r) o[r] = cacc[jd][r] / lrow[r];
        const int c = h * DD + jd * 16 + nl;
        const size_t base = (b * CC + c) * (size_t)SS + q0 + wave * 16 + mh;
        float4 o0 = make_float4(o[0], o[1], o[2], o[3]);
        float4 o1 = make_float4(o[4], o[5], o[6], o[7]);
        *(float4*)(ctx + base)     = o0;
        *(float4*)(ctx + base + 4) = o1;
    }
}

// ---------------------------------------------------------------------------
// LayerNorm over channel dim of NCW, in-place. grid = (S/64, B), block = 256.
// Thread (sl, cq): owns s-column sl, reduces c = cq, cq+4, ...
// ---------------------------------------------------------------------------
__global__ __launch_bounds__(256) void ln_nchw(
    float* __restrict__ x, const float* __restrict__ g,
    const float* __restrict__ bt, int C_, int N_)
{
    __shared__ float rsum[256], rsq[256];
    __shared__ float smu[64], srs[64];
    const int t = threadIdx.x, sl = t & 63, cq = t >> 6;
    const size_t b = blockIdx.y;
    const int s0 = blockIdx.x * 64;

    float sum = 0.f, sq = 0.f;
    for (int c = cq; c < C_; c += 4) {
        const float v = x[(b * C_ + c) * (size_t)N_ + s0 + sl];
        sum += v; sq += v * v;
    }
    rsum[t] = sum; rsq[t] = sq;
    __syncthreads();
    if (cq == 0) {
        const float s1 = rsum[sl] + rsum[64 + sl] + rsum[128 + sl] + rsum[192 + sl];
        const float s2 = rsq[sl] + rsq[64 + sl] + rsq[128 + sl] + rsq[192 + sl];
        const float mu = s1 / (float)C_;
        const float var = s2 / (float)C_ - mu * mu;
        smu[sl] = mu;
        srs[sl] = rsqrtf(var + 1e-12f);
    }
    __syncthreads();
    const float mu = smu[sl], rs = srs[sl];
    for (int c = cq; c < C_; c += 4) {
        const size_t idx = (b * C_ + c) * (size_t)N_ + s0 + sl;
        x[idx] = (x[idx] - mu) * rs * g[c] + bt[c];
    }
}

// ---------------------------------------------------------------------------
extern "C" void kernel_launch(void* const* d_in, const int* in_sizes, int n_in,
                              void* d_out, int out_size, void* d_ws, size_t ws_size,
                              hipStream_t stream) {
    (void)in_sizes; (void)n_in; (void)out_size; (void)ws_size;
    const float* x       = (const float*)d_in[0];
    const float* mask    = (const float*)d_in[1];
    const float* Wq      = (const float*)d_in[2];
    const float* bq      = (const float*)d_in[3];
    const float* Wk      = (const float*)d_in[4];
    const float* bk      = (const float*)d_in[5];
    const float* Wv      = (const float*)d_in[6];
    const float* bv      = (const float*)d_in[7];
    const float* Wpost   = (const float*)d_in[8];
    const float* bpost   = (const float*)d_in[9];
    const float* gpost   = (const float*)d_in[10];
    const float* betapost= (const float*)d_in[11];
    const float* Wint    = (const float*)d_in[12];
    const float* bint    = (const float*)d_in[13];
    const float* Wout    = (const float*)d_in[14];
    const float* bout    = (const float*)d_in[15];
    const float* gout    = (const float*)d_in[16];
    const float* betaout = (const float*)d_in[17];
    float* out = (float*)d_out;

    const size_t SZ = (size_t)BB * CC * SS;      // 12.58M floats = 48 MiB
    float* qbuf   = (float*)d_ws;
    float* kbuf   = qbuf + SZ;
    float* vbuf   = kbuf + SZ;
    float* postb  = vbuf + SZ;
    float* interb = postb + SZ;                  // 4 batches of I*S == SZ elems

    const dim3 blk(256);

    // Q,K,V 1x1 convs (per-batch GEMM M=768,N=1024,K=768)
    wmma_gemm_nchw<false, false><<<dim3(8, 6, 16), blk, 0, stream>>>(x, Wq, bq, nullptr, qbuf, CC, SS, CC);
    wmma_gemm_nchw<false, false><<<dim3(8, 6, 16), blk, 0, stream>>>(x, Wk, bk, nullptr, kbuf, CC, SS, CC);
    wmma_gemm_nchw<false, false><<<dim3(8, 6, 16), blk, 0, stream>>>(x, Wv, bv, nullptr, vbuf, CC, SS, CC);

    // Flash attention; ctx aliases qbuf (each WG overwrites only the Q region
    // it alone reads, after its final Q read).
    attn_flash<<<dim3(SS / 64, HH, BB), dim3(128), 0, stream>>>(qbuf, kbuf, vbuf, mask, qbuf);

    // post conv + residual(x), then LN in-place
    wmma_gemm_nchw<true, false><<<dim3(8, 6, 16), blk, 0, stream>>>(qbuf, Wpost, bpost, x, postb, CC, SS, CC);
    ln_nchw<<<dim3(SS / 64, BB), blk, 0, stream>>>(postb, gpost, betapost, CC, SS);

    // FFN in 4 batch-chunks of 4 (keeps inter at 48 MiB)
    for (int cb = 0; cb < 4; ++cb) {
        const float* pin = postb + (size_t)cb * 4 * CC * SS;
        float* o = out + (size_t)cb * 4 * CC * SS;
        wmma_gemm_nchw<false, true><<<dim3(8, II / 128, 4), blk, 0, stream>>>(pin, Wint, bint, nullptr, interb, II, SS, CC);
        wmma_gemm_nchw<true, false><<<dim3(8, CC / 128, 4), blk, 0, stream>>>(interb, Wout, bout, pin, o, CC, SS, II);
    }
    // final LN in-place on d_out
    ln_nchw<<<dim3(SS / 64, BB), blk, 0, stream>>>(out, gout, betaout, CC, SS);
}
